// VQVAE_32538672234737
// MI455X (gfx1250) — matrix operand order repaired and checked by reading, not compiled, not run on hardware
//
#include <hip/hip_runtime.h>
#include <hip/hip_bf16.h>
#include <math.h>

// ---------------------------------------------------------------------------
// VQ-VAE forward on gfx1250 (CDNA5, wave32, WMMA + TDM).
// All convs / deconvs are implicit-GEMM with v_wmma_f32_16x16x32_f16.
// Activations flow in f16 NHWC; weights prepacked to f16 [Cout][KH][KW][Cin].
// B (weight) tiles are DMA'd global->LDS by the Tensor Data Mover
// (tensor_load_to_lds + s_wait_tensorcnt), with hardware OOB zero-fill.
// Deconvs = fractionally-strided convs (lhs_dilation=2, flipped weights).
// VQ = WMMA GEMM (z @ emb^T) with fused argmin + gather + loss reduction.
// ---------------------------------------------------------------------------

typedef __attribute__((ext_vector_type(16))) _Float16 v16h;
typedef __attribute__((ext_vector_type(8)))  _Float16 v8h;
typedef __attribute__((ext_vector_type(8)))  float    v8f;
typedef __attribute__((ext_vector_type(4)))  unsigned int u32x4;
typedef __attribute__((ext_vector_type(8)))  int          i32x8;
typedef __attribute__((ext_vector_type(4)))  int          i32x4;

union U16h { v16h v; _Float16 e[16]; };

// ---------------------------------------------------------------------------
// Generic implicit-GEMM conv kernel.
//  MODE 0: out = relu(conv+bias), stored f16 NHWC
//  MODE 1: out = sigmoid(conv+bias), stored f32 NCHW (final layer -> d_out)
// Block: 128 threads (4 waves). Tile: 64 (M) x 64 (N). K-step: 32.
// Wave w owns rows [16w,16w+16); loops 4 N-subtiles -> 4 f32 accumulators.
// B tile fetched by TDM (wave 0 issues; TENSORcnt + barrier synchronize).
// ---------------------------------------------------------------------------
template <int MODE>
__global__ __launch_bounds__(128)
void conv_gemm(const _Float16* __restrict__ in, const _Float16* __restrict__ wgt,
               const float* __restrict__ bias, void* __restrict__ outp,
               int Bn, int IH, int IW, int Cin,
               int OH, int OW, int Cout,
               int KH, int KW, int stride, int pad, int ldil, int Ktot)
{
    __shared__ _Float16 As[64 * 32];   // A tile: [row][k]
    __shared__ _Float16 Bs[64 * 32];   // B^T tile: [n][k]  (TDM destination)

    const int tid  = threadIdx.x;
    const int wave = tid >> 5;
    const int lane = tid & 31;
    const int half = lane >> 4;
    const int l16  = lane & 15;

    const int mbase = blockIdx.x * 64;
    const int nbase = blockIdx.y * 64;
    const int M     = Bn * OH * OW;
    const int KWC   = KW * Cin;
    const int OHW   = OH * OW;

    // ---- hoisted per-thread A-loader coordinates (K-invariant) -------------
    // Thread loads row (tid>>1), 16-wide k-chunk at (tid&1)*16 each K-step.
    const int arow = tid >> 1;
    const int akc  = (tid & 1) * 16;
    const int am   = mbase + arow;
    const bool amok = (am < M);
    int ab = 0, aoh = 0, aow = 0;
    if (amok) {
        ab = am / OHW;
        int r = am - ab * OHW;
        aoh = r / OW;
        aow = r - aoh * OW;
    }
    const bool vecPath = ((Cin & 15) == 0);   // 16-chunk = 16 contiguous channels

    v8f acc[4];
#pragma unroll
    for (int i = 0; i < 4; ++i) acc[i] = (v8f){};

    for (int k0 = 0; k0 < Ktot; k0 += 32) {
        // ---- B^T tile via Tensor Data Mover (wave 0 only) ------------------
        if (wave == 0) {
            unsigned int ldsB = (unsigned int)(uintptr_t)(&Bs[0]); // LDS byte addr
            unsigned long long ga =
                (unsigned long long)(uintptr_t)(wgt + (size_t)nbase * Ktot + k0);
            int kRem = Ktot - k0;   if (kRem > 32) kRem = 32;   // OOB -> zero
            int nRem = Cout - nbase; if (nRem > 64) nRem = 64;

            u32x4 g0;
            g0[0] = 1u;                                  // count=1, user D#
            g0[1] = ldsB;                                // lds_addr
            g0[2] = (unsigned int)(ga & 0xFFFFFFFFu);    // global_addr lo
            g0[3] = (unsigned int)((ga >> 32) & 0x1FFFFFFu) | (2u << 30); // hi + type=2

            i32x8 g1;
            g1[0] = (int)(1u << 16);                     // data_size = 2 bytes
            g1[1] = (int)(((unsigned int)kRem & 0xFFFFu) << 16);          // dim0 lo16
            g1[2] = (int)((((unsigned int)kRem >> 16) & 0xFFFFu)
                          | (((unsigned int)nRem & 0xFFFFu) << 16));      // dim0 hi, dim1 lo
            g1[3] = (int)((((unsigned int)nRem >> 16) & 0xFFFFu)
                          | (32u << 16));                // dim1 hi, tile_dim0 = 32
            g1[4] = (int)64u;                            // tile_dim1 = 64, tile_dim2 = 0
            g1[5] = (int)(unsigned int)Ktot;             // tensor_dim0_stride lo
            g1[6] = 0;                                   // stride0 hi, stride1 lo
            g1[7] = 0;                                   // stride1 hi

            i32x4 z4 = {0, 0, 0, 0};
            i32x8 z8 = {0, 0, 0, 0, 0, 0, 0, 0};
            __builtin_amdgcn_tensor_load_to_lds(g0, g1, z4, z4, z8, 0);
        }

        // ---- A tile: implicit im2col gather (f16 in, NHWC) -----------------
        if (vecPath) {
            v8h lo = (v8h){}, hi = (v8h){};
            if (amok) {
                int k  = k0 + akc;                 // chunk within one (kh,kw)
                int kh = k / KWC;
                int r2 = k - kh * KWC;
                int kw = r2 / Cin;
                int ci = r2 - kw * Cin;
                int iy = aoh * stride - pad + kh;
                int ix = aow * stride - pad + kw;
                bool ok = (iy >= 0) && (ix >= 0);
                if (ok && ldil > 1) {
                    ok = ((iy % ldil) == 0) && ((ix % ldil) == 0);
                    iy /= ldil; ix /= ldil;
                }
                if (ok && iy < IH && ix < IW) {
                    const _Float16* src =
                        in + (((size_t)ab * IH + iy) * IW + ix) * Cin + ci;
                    lo = *(const v8h*)src;
                    hi = *(const v8h*)(src + 8);
                }
            }
            *(v8h*)&As[arow * 32 + akc]     = lo;
            *(v8h*)&As[arow * 32 + akc + 8] = hi;
        } else {
#pragma unroll 4
            for (int j = 0; j < 16; ++j) {
                int k = k0 + akc + j;
                _Float16 val = (_Float16)0.f;
                if (amok && k < Ktot) {
                    int kh = k / KWC;
                    int r2 = k - kh * KWC;
                    int kw = r2 / Cin;
                    int ci = r2 - kw * Cin;
                    int iy = aoh * stride - pad + kh;
                    int ix = aow * stride - pad + kw;
                    bool ok = (iy >= 0) && (ix >= 0);
                    if (ok && ldil > 1) {
                        ok = ((iy % ldil) == 0) && ((ix % ldil) == 0);
                        iy /= ldil; ix /= ldil;
                    }
                    if (ok && iy < IH && ix < IW)
                        val = in[(((size_t)ab * IH + iy) * IW + ix) * Cin + ci];
                }
                As[arow * 32 + akc + j] = val;
            }
        }

        if (wave == 0) __builtin_amdgcn_s_wait_tensorcnt(0);
        __syncthreads();

        // ---- fragments + WMMA ---------------------------------------------
        // A frag: row = wave*16 + l16; K = half*16 + 0..15 (contiguous row)
        U16h a;
        {
            const _Float16* ap = &As[(wave * 16 + l16) * 32 + half * 16];
            v8h lo = *(const v8h*)ap;
            v8h hi = *(const v8h*)(ap + 8);
#pragma unroll
            for (int i = 0; i < 8; ++i) { a.e[i] = lo[i]; a.e[8 + i] = hi[i]; }
        }
#pragma unroll
        for (int nt = 0; nt < 4; ++nt) {
            // B frag: col n = nt*16 + l16; K = half*16 + 0..15 (contiguous row)
            U16h bf;
            const _Float16* bp = &Bs[(nt * 16 + l16) * 32 + half * 16];
            v8h lo = *(const v8h*)bp;
            v8h hi = *(const v8h*)(bp + 8);
#pragma unroll
            for (int i = 0; i < 8; ++i) { bf.e[i] = lo[i]; bf.e[8 + i] = hi[i]; }
            acc[nt] = __builtin_amdgcn_wmma_f32_16x16x32_f16(
                false, a.v, false, bf.v, (short)0, acc[nt], false, false);
        }
        __syncthreads();
    }

    // ---- epilogue: C layout -> VGPR v, lane: M = v + 8*half, N = l16 -------
#pragma unroll
    for (int nt = 0; nt < 4; ++nt) {
#pragma unroll
        for (int v = 0; v < 8; ++v) {
            int m = mbase + wave * 16 + v + 8 * half;
            int n = nbase + nt * 16 + l16;
            if (m >= M || n >= Cout) continue;
            float val = acc[nt][v] + bias[n];
            if (MODE == 0) {
                val = val > 0.f ? val : 0.f;
                ((_Float16*)outp)[(size_t)m * Cout + n] = (_Float16)val;
            } else {
                val = 1.f / (1.f + __expf(-val));
                int b  = m / OHW;
                int r  = m - b * OHW;
                int oh = r / OW;
                int ow = r - oh * OW;
                ((float*)outp)[(((size_t)b * Cout + n) * OH + oh) * OW + ow] = val;
            }
        }
    }
}

// ---------------------------------------------------------------------------
// Vector quantization: one block = 16 rows of z (D=64), all 512 codes.
// argmin_n ( ||e_n||^2 - 2 z.e_n ) via WMMA; gather q; accumulate loss sum.
// ---------------------------------------------------------------------------
__global__ __launch_bounds__(128)
void vq_kernel(const _Float16* __restrict__ z, const _Float16* __restrict__ emb16,
               const float* __restrict__ enorm, _Float16* __restrict__ q,
               float* __restrict__ lossAcc, int M)
{
    __shared__ _Float16 Zs[16 * 64];
    __shared__ float    redV[4 * 32 * 8];
    __shared__ int      redN[4 * 32 * 8];
    __shared__ int      rowIdx[16];
    __shared__ float    redS[128];

    const int tid  = threadIdx.x;
    const int wave = tid >> 5;
    const int lane = tid & 31;
    const int half = lane >> 4;
    const int l16  = lane & 15;
    const int mbase = blockIdx.x * 16;

    for (int i = tid; i < 16 * 64; i += 128) {
        int row = i >> 6;
        Zs[i] = (mbase + row < M) ? z[(size_t)(mbase + row) * 64 + (i & 63)]
                                  : (_Float16)0.f;
    }
    __syncthreads();

    // A fragments for the two K-steps (shared by all 8 subtiles of this wave)
    U16h a0, a1;
#pragma unroll
    for (int i = 0; i < 8; ++i) {
        a0.e[i]     = Zs[l16 * 64 +  0 + half * 16 + i];
        a0.e[8 + i] = Zs[l16 * 64 +  8 + half * 16 + i];
        a1.e[i]     = Zs[l16 * 64 + 32 + half * 16 + i];
        a1.e[8 + i] = Zs[l16 * 64 + 40 + half * 16 + i];
    }

    float bestv[8];
    int   bestn[8];
#pragma unroll
    for (int v = 0; v < 8; ++v) { bestv[v] = 3.4e38f; bestn[v] = 0; }

    for (int nt = 0; nt < 8; ++nt) {
        int gn = wave * 128 + nt * 16 + l16;       // 0..511
        v8f acc = (v8f){};
#pragma unroll
        for (int ks = 0; ks < 2; ++ks) {
            U16h bf;
            const _Float16* bp = emb16 + (size_t)gn * 64 + ks * 32 + half * 16;
            v8h lo = *(const v8h*)bp;
            v8h hi = *(const v8h*)(bp + 8);
#pragma unroll
            for (int i = 0; i < 8; ++i) { bf.e[i] = lo[i]; bf.e[8 + i] = hi[i]; }
            acc = __builtin_amdgcn_wmma_f32_16x16x32_f16(
                false, (ks ? a1.v : a0.v), false, bf.v, (short)0, acc, false, false);
        }
        float en = enorm[gn];
#pragma unroll
        for (int v = 0; v < 8; ++v) {
            float d = en - 2.f * acc[v];           // row = v + 8*half
            if (d < bestv[v]) { bestv[v] = d; bestn[v] = gn; }
        }
    }

#pragma unroll
    for (int v = 0; v < 8; ++v) {
        redV[(wave * 32 + lane) * 8 + v] = bestv[v];
        redN[(wave * 32 + lane) * 8 + v] = bestn[v];
    }
    __syncthreads();

    if (tid < 16) {                // per-row final argmin over 64 candidates
        int r = tid, v = r & 7, hg = r >> 3;
        float bv = 3.4e38f; int bn = 0;
        for (int w = 0; w < 4; ++w)
            for (int l = 0; l < 16; ++l) {
                int sl = w * 32 + hg * 16 + l;
                float val = redV[sl * 8 + v];
                if (val < bv) { bv = val; bn = redN[sl * 8 + v]; }
            }
        rowIdx[r] = bn;
    }
    __syncthreads();

    float part = 0.f;
    for (int i = tid; i < 16 * 64; i += 128) {
        int row = i >> 6, dd = i & 63;
        if (mbase + row < M) {
            _Float16 qv = emb16[(size_t)rowIdx[row] * 64 + dd];
            float diff = (float)qv - (float)Zs[i];
            part += diff * diff;
            q[(size_t)(mbase + row) * 64 + dd] = qv;
        }
    }
    redS[tid] = part;
    __syncthreads();
    if (tid == 0) {
        float s = 0.f;
        for (int i = 0; i < 128; ++i) s += redS[i];
        atomicAdd(lossAcc, s);
    }
}

// ---------------------------------------------------------------------------
// Prep / layout kernels
// ---------------------------------------------------------------------------
__global__ void nchw_to_nhwc_f16(const float* __restrict__ x, _Float16* __restrict__ o,
                                 int Bn, int C, int H, int W)
{
    int i = blockIdx.x * blockDim.x + threadIdx.x;
    int tot = Bn * C * H * W;
    if (i >= tot) return;
    int ci = i % C; int r = i / C;
    int w  = r % W; r /= W;
    int h  = r % H; int b = r / H;
    o[i] = (_Float16)x[(((size_t)b * C + ci) * H + h) * W + w];
}

// OIHW f32 -> [O][KH][KW][I] f16
__global__ void pack_conv_w(const float* __restrict__ w, _Float16* __restrict__ o,
                            int Cout, int Cin, int KH, int KW)
{
    int i = blockIdx.x * blockDim.x + threadIdx.x;
    int tot = Cout * Cin * KH * KW;
    if (i >= tot) return;
    int ci = i % Cin; int r = i / Cin;
    int kw = r % KW;  r /= KW;
    int kh = r % KH;  int oc = r / KH;
    o[i] = (_Float16)w[(((size_t)oc * Cin + ci) * KH + kh) * KW + kw];
}

// [I][O][KH][KW] f32 (ConvTranspose) -> flipped [O][KH][KW][I] f16
__global__ void pack_deconv_w(const float* __restrict__ w, _Float16* __restrict__ o,
                              int Cin, int Cout, int KH, int KW)
{
    int i = blockIdx.x * blockDim.x + threadIdx.x;
    int tot = Cout * Cin * KH * KW;
    if (i >= tot) return;
    int ci = i % Cin; int r = i / Cin;
    int kw = r % KW;  r /= KW;
    int kh = r % KH;  int oc = r / KH;
    o[i] = (_Float16)w[(((size_t)ci * Cout + oc) * KH + (KH - 1 - kh)) * KW
                       + (KW - 1 - kw)];
}

__global__ void prep_emb(const float* __restrict__ emb, _Float16* __restrict__ emb16,
                         float* __restrict__ enorm)
{
    __shared__ float s[64];
    int n = blockIdx.x, t = threadIdx.x;          // 512 x 64
    float v = emb[(size_t)n * 64 + t];
    emb16[(size_t)n * 64 + t] = (_Float16)v;
    s[t] = v * v;
    __syncthreads();
    if (t == 0) {
        float a = 0.f;
        for (int i = 0; i < 64; ++i) a += s[i];
        enorm[n] = a;
    }
}

__global__ void zero_f32(float* p) { p[0] = 0.f; }

__global__ void finalize_loss(const float* __restrict__ acc, float* __restrict__ out)
{
    // loss = mean((q-z)^2) + 0.25*mean((q-z)^2), mean over 32*64*64*64 elems
    out[0] = 1.25f * acc[0] / 8388608.0f;
}

// ---------------------------------------------------------------------------
extern "C" void kernel_launch(void* const* d_in, const int* in_sizes, int n_in,
                              void* d_out, int out_size, void* d_ws, size_t ws_size,
                              hipStream_t stream)
{
    const float* x   = (const float*)d_in[0];
    const float* w1  = (const float*)d_in[1];  const float* b1  = (const float*)d_in[2];
    const float* w2  = (const float*)d_in[3];  const float* b2  = (const float*)d_in[4];
    const float* w3  = (const float*)d_in[5];  const float* b3  = (const float*)d_in[6];
    const float* emb = (const float*)d_in[7];
    const float* dw1 = (const float*)d_in[8];  const float* db1 = (const float*)d_in[9];
    const float* dw2 = (const float*)d_in[10]; const float* db2 = (const float*)d_in[11];
    const float* dw3 = (const float*)d_in[12]; const float* db3 = (const float*)d_in[13];

    // ---- workspace layout (ping-pong activation regions + small buffers) ---
    char* p = (char*)d_ws;
    auto alloc = [&](size_t bytes) -> char* {
        char* r = p; p += (bytes + 255) & ~(size_t)255; return r;
    };
    _Float16* regA  = (_Float16*)alloc(268435456ull);  // xh / h2 / q / d2o
    _Float16* regB  = (_Float16*)alloc(134217728ull);  // h1 / z / d1o
    _Float16* w1p   = (_Float16*)alloc(64 * 48 * 2);
    _Float16* w2p   = (_Float16*)alloc(128 * 1024 * 2);
    _Float16* w3p   = (_Float16*)alloc(64 * 1152 * 2);
    _Float16* d1p   = (_Float16*)alloc(128 * 1024 * 2);
    _Float16* d2p   = (_Float16*)alloc(64 * 2048 * 2);
    _Float16* d3p   = (_Float16*)alloc(3 * 576 * 2);
    _Float16* emb16 = (_Float16*)alloc(512 * 64 * 2);
    float*    enorm = (float*)   alloc(512 * 4);
    float*    lossA = (float*)   alloc(256);

    // ---- prep ----
    {
        int tot = 32 * 3 * 256 * 256;
        nchw_to_nhwc_f16<<<(tot + 255) / 256, 256, 0, stream>>>(x, regA, 32, 3, 256, 256);
    }
    pack_conv_w<<<(64 * 3 * 16   + 255) / 256, 256, 0, stream>>>(w1, w1p, 64, 3, 4, 4);
    pack_conv_w<<<(128 * 64 * 16 + 255) / 256, 256, 0, stream>>>(w2, w2p, 128, 64, 4, 4);
    pack_conv_w<<<(64 * 128 * 9  + 255) / 256, 256, 0, stream>>>(w3, w3p, 64, 128, 3, 3);
    pack_deconv_w<<<(128 * 64 * 16 + 255) / 256, 256, 0, stream>>>(dw1, d1p, 64, 128, 4, 4);
    pack_deconv_w<<<(64 * 128 * 16 + 255) / 256, 256, 0, stream>>>(dw2, d2p, 128, 64, 4, 4);
    pack_deconv_w<<<(3 * 64 * 9    + 255) / 256, 256, 0, stream>>>(dw3, d3p, 64, 3, 3, 3);
    prep_emb<<<512, 64, 0, stream>>>(emb, emb16, enorm);
    zero_f32<<<1, 1, 0, stream>>>(lossA);

    // ---- encoder ----
    // L1: (256,256,3) -> (128,128,64), k4 s2 p1, K=48
    conv_gemm<0><<<dim3(524288 / 64, 1), 128, 0, stream>>>(
        regA, w1p, b1, regB, 32, 256, 256, 3, 128, 128, 64, 4, 4, 2, 1, 1, 48);
    // L2: (128,128,64) -> (64,64,128), k4 s2 p1, K=1024
    conv_gemm<0><<<dim3(131072 / 64, 2), 128, 0, stream>>>(
        regB, w2p, b2, regA, 32, 128, 128, 64, 64, 64, 128, 4, 4, 2, 1, 1, 1024);
    // L3: (64,64,128) -> z (64,64,64), k3 s1 p1, K=1152
    conv_gemm<0><<<dim3(131072 / 64, 1), 128, 0, stream>>>(
        regA, w3p, b3, regB, 32, 64, 64, 128, 64, 64, 64, 3, 3, 1, 1, 1, 1152);

    // ---- vector quantization (z in regB -> q in regA) ----
    vq_kernel<<<131072 / 16, 128, 0, stream>>>(regB, emb16, enorm, regA, lossA, 131072);

    // ---- decoder (deconvs as lhs-dilated convs) ----
    // D1: (64,64,64) -> (128,128,128), k4 ldil2 p_eff=2, K=1024
    conv_gemm<0><<<dim3(524288 / 64, 2), 128, 0, stream>>>(
        regA, d1p, db1, regB, 32, 64, 64, 64, 128, 128, 128, 4, 4, 1, 2, 2, 1024);
    // D2: (128,128,128) -> (256,256,64), k4 ldil2 p_eff=2, K=2048
    conv_gemm<0><<<dim3(2097152 / 64, 1), 128, 0, stream>>>(
        regB, d2p, db2, regA, 32, 128, 128, 128, 256, 256, 64, 4, 4, 1, 2, 2, 2048);
    // D3: (256,256,64) -> (256,256,3) sigmoid, f32 NCHW into d_out, K=576
    conv_gemm<1><<<dim3(2097152 / 64, 1), 128, 0, stream>>>(
        regA, d3p, db3, d_out, 32, 256, 256, 64, 256, 256, 3, 3, 3, 1, 1, 1, 576);

    // ---- loss scalar (last element of d_out) ----
    finalize_loss<<<1, 1, 0, stream>>>(lossA, (float*)d_out + (out_size - 1));

    (void)in_sizes; (void)n_in; (void)ws_size;
}
